// GraphAttention_52338471469220
// MI455X (gfx1250) — matrix operand order repaired and checked by reading, not compiled, not run on hardware
//
#include <hip/hip_runtime.h>

#define N_NODES 8192
#define IN_F    512
#define OUT_F   64
#define ALPHA   0.2f
#define NSPLIT  4
#define JCHUNK  (N_NODES / NSPLIT)   // 2048 columns per j-split
#define NBLK    (JCHUNK / 32)        // 64 j-blocks per split

typedef __attribute__((ext_vector_type(16))) _Float16 v16h;
typedef __attribute__((ext_vector_type(8)))  _Float16 v8h;
typedef __attribute__((ext_vector_type(8)))  float    v8f;
typedef __attribute__((ext_vector_type(4)))  int      vi4;
typedef __attribute__((ext_vector_type(4)))  float    vf4;

union AccT  { v8f  v; float    e[8];  };
union AFrag { v16h v; _Float16 e[16]; };
union BFrag { v16h v; v8h      h[2];  };

// gfx1250 async global->LDS path (ASYNCcnt-tracked), guarded so the file
// compiles on toolchains that don't declare the builtins.
#if defined(__has_builtin)
#if __has_builtin(__builtin_amdgcn_global_load_async_to_lds_b128) && \
    __has_builtin(__builtin_amdgcn_global_load_async_to_lds_b32)  && \
    __has_builtin(__builtin_amdgcn_s_wait_asynccnt)
#define GAT_ASYNC_LDS 1
#endif
#endif

#ifdef GAT_ASYNC_LDS
// Builtin params are (AS1 int-vector*/int* src, AS3 int-vector*/int* dst, imm, imm).
// Build the address-space pointers via integer casts (no implicit AS conversion).
#define AS_G_V4(p)  ((__attribute__((address_space(1))) vi4*)(unsigned long long)(const void*)(p))
#define AS_L_V4(p)  ((__attribute__((address_space(3))) vi4*)(unsigned)(unsigned long long)(void*)(p))
#define AS_G_I32(p) ((__attribute__((address_space(1))) int*)(unsigned long long)(const void*)(p))
#define AS_L_I32(p) ((__attribute__((address_space(3))) int*)(unsigned)(unsigned long long)(void*)(p))
#endif

// ---------------------------------------------------------------------------
// Kernel 1: Wh = h @ W   (8192x512 @ 512x64), f16 WMMA, fp32 accumulate.
// Writes Wh (f32 row-major) and WhT (f16 [64][8192] column-major) so kernel 2
// can fetch B-fragments with contiguous 32B reads.
// ---------------------------------------------------------------------------
__global__ __launch_bounds__(128)
void gat_wh_gemm(const float* __restrict__ h, const float* __restrict__ W,
                 float* __restrict__ Wh, _Float16* __restrict__ WhT) {
  // Transposed f16 copy of W in LDS: ldsWT[col][k], row stride 520 halves
  // (1040B) -> b128 reads land on distinct bank groups.
  __shared__ _Float16 ldsWT[64 * 520];
  const int t = threadIdx.x;
  {
    const int col = t & 63;
    const int ks  = (t >> 6) * 256;
    #pragma unroll 4
    for (int k = 0; k < 256; ++k)
      ldsWT[col * 520 + ks + k] = (_Float16)W[(ks + k) * 64 + col];
  }
  __syncthreads();

  const int wave = t >> 5, l = t & 31;
  const int half = l >> 4, lm = l & 15, koff = half * 8;
  const int rowbase = blockIdx.x * 64 + wave * 16;
  const int arow = rowbase + lm;            // A-fragment: lane holds row l%16

  AccT c[4];
  #pragma unroll
  for (int i = 0; i < 4; ++i)
    #pragma unroll
    for (int j = 0; j < 8; ++j) c[i].e[j] = 0.0f;

  for (int ks = 0; ks < IN_F; ks += 32) {
    const float* hp = h + (long)arow * IN_F + ks + koff;
    // h is streamed exactly once -> non-temporal
    const vf4 h0 = __builtin_nontemporal_load((const vf4*)(hp));
    const vf4 h1 = __builtin_nontemporal_load((const vf4*)(hp + 4));
    const vf4 h2 = __builtin_nontemporal_load((const vf4*)(hp + 16));
    const vf4 h3 = __builtin_nontemporal_load((const vf4*)(hp + 20));
    AFrag A;
    A.e[0]=(_Float16)h0.x; A.e[1]=(_Float16)h0.y; A.e[2]=(_Float16)h0.z; A.e[3]=(_Float16)h0.w;
    A.e[4]=(_Float16)h1.x; A.e[5]=(_Float16)h1.y; A.e[6]=(_Float16)h1.z; A.e[7]=(_Float16)h1.w;
    A.e[8]=(_Float16)h2.x; A.e[9]=(_Float16)h2.y; A.e[10]=(_Float16)h2.z; A.e[11]=(_Float16)h2.w;
    A.e[12]=(_Float16)h3.x; A.e[13]=(_Float16)h3.y; A.e[14]=(_Float16)h3.z; A.e[15]=(_Float16)h3.w;
    #pragma unroll
    for (int nt = 0; nt < 4; ++nt) {
      const _Float16* bp = &ldsWT[(nt * 16 + lm) * 520 + ks + half * 8];
      BFrag B;
      B.h[0] = *(const v8h*)bp;
      B.h[1] = *(const v8h*)(bp + 16);
      c[nt].v = __builtin_amdgcn_wmma_f32_16x16x32_f16(
          false, A.v, false, B.v, (short)0, c[nt].v, false, false);
    }
  }

  #pragma unroll
  for (int nt = 0; nt < 4; ++nt)
    #pragma unroll
    for (int r = 0; r < 8; ++r) {
      const int rr = rowbase + r + half * 8;   // C layout: VGPR r -> M=r / r+8
      const int cc = nt * 16 + lm;
      const float val = c[nt].e[r];
      Wh[rr * OUT_F + cc] = val;
      WhT[(long)cc * N_NODES + rr] = (_Float16)val;
    }
}

// ---------------------------------------------------------------------------
// Kernel 1b: Wh1 = Wh @ a[:64], Wh2 = Wh @ a[64:], fp32 exact.
// ---------------------------------------------------------------------------
__global__ __launch_bounds__(256)
void gat_wh12(const float* __restrict__ Wh, const float* __restrict__ a,
              float* __restrict__ wh1, float* __restrict__ wh2) {
  const int row = blockIdx.x * 256 + threadIdx.x;
  const float* wr = Wh + row * OUT_F;
  float s1 = 0.0f, s2 = 0.0f;
  #pragma unroll 8
  for (int c = 0; c < OUT_F; ++c) {
    const float w = wr[c];
    s1 += w * a[c];
    s2 += w * a[OUT_F + c];
  }
  wh1[row] = s1;
  wh2[row] = s2;
}

// ---------------------------------------------------------------------------
// Kernel 2: fused mask + leaky-relu + exp + (P @ Wh) accumulation.
// Bounded logits -> no online max; exp(e) fits f16/f32 comfortably.
// Double-buffered LDS staging (async-to-LDS, ASYNCcnt-tracked), one barrier
// per 32-j block; adj streamed once with non-temporal b128.
// ---------------------------------------------------------------------------
__global__ __launch_bounds__(128)
void gat_attn(const int* __restrict__ adj, const _Float16* __restrict__ WhT,
              const float* __restrict__ wh1v, const float* __restrict__ wh2v,
              float* __restrict__ pacc, float* __restrict__ pl) {
  __shared__ _Float16 ldsB[2][64 * 40];   // [col][32 j] padded to 40 halves
  __shared__ float    ldsW2[2][32];
  const int t = threadIdx.x;
  const int wave = t >> 5, l = t & 31;
  const int half = l >> 4, lm = l & 15, koff = half * 8;
  const int row0  = blockIdx.x * 64;
  const int myrow = row0 + wave * 16 + lm;   // A-fragment row for this lane
  const float w1  = wh1v[myrow];
  const int jstart = blockIdx.y * JCHUNK;
  const int scol = t >> 1, sseg = t & 1;     // staging assignment

  AccT c[4];
  #pragma unroll
  for (int i = 0; i < 4; ++i)
    #pragma unroll
    for (int j = 0; j < 8; ++j) c[i].e[j] = 0.0f;
  float lsum = 0.0f;

  auto stage = [&](int buf, int jb) {
    const _Float16* src = WhT + (long)scol * N_NODES + jb + sseg * 16;
    _Float16* dst = &ldsB[buf][scol * 40 + sseg * 16];
#ifdef GAT_ASYNC_LDS
    __builtin_amdgcn_global_load_async_to_lds_b128(AS_G_V4(src), AS_L_V4(dst), 0, 0);
    __builtin_amdgcn_global_load_async_to_lds_b128(AS_G_V4(src + 8), AS_L_V4(dst + 8), 0, 0);
    if (t < 32)
      __builtin_amdgcn_global_load_async_to_lds_b32(AS_G_I32(wh2v + jb + t),
                                                    AS_L_I32(&ldsW2[buf][t]), 0, 0);
#else
    *(v8h*)dst       = *(const v8h*)src;
    *(v8h*)(dst + 8) = *(const v8h*)(src + 8);
    if (t < 32) ldsW2[buf][t] = wh2v[jb + t];
#endif
  };

  auto pv = [&](int av, float w2) -> float {
    const float x = w1 + w2;
    const float s = x > 0.0f ? x : ALPHA * x;   // leaky relu
    const float p = __expf(s);
    return (av != 0) ? p : 0.0f;                // mask -> exp(NEG_BIG) == 0
  };

  stage(0, jstart);
  for (int i = 0; i < NBLK; ++i) {
    const int jb  = jstart + i * 32;
    const int buf = i & 1;
#ifdef GAT_ASYNC_LDS
    __builtin_amdgcn_s_wait_asynccnt(0);   // drain this wave's async stages
#endif
    __syncthreads();                       // publish buf, retire prior reads
    if (i + 1 < NBLK) stage(buf ^ 1, jb + 32);

    // adj: 16 ints per lane in A-fragment K order (koff..+7, koff+16..+23);
    // 256 MB streamed once -> non-temporal (don't thrash 192 MB L2).
    const int* ap = adj + (long)myrow * N_NODES + jb + koff;
    const vi4 a0 = __builtin_nontemporal_load((const vi4*)(ap));
    const vi4 a1 = __builtin_nontemporal_load((const vi4*)(ap + 4));
    const vi4 a2 = __builtin_nontemporal_load((const vi4*)(ap + 16));
    const vi4 a3 = __builtin_nontemporal_load((const vi4*)(ap + 20));
    const float4 w20 = *(const float4*)&ldsW2[buf][koff];
    const float4 w21 = *(const float4*)&ldsW2[buf][koff + 4];
    const float4 w22 = *(const float4*)&ldsW2[buf][koff + 16];
    const float4 w23 = *(const float4*)&ldsW2[buf][koff + 20];

    AFrag P;
    float p;
    p = pv(a0.x, w20.x); lsum += p; P.e[0]  = (_Float16)p;
    p = pv(a0.y, w20.y); lsum += p; P.e[1]  = (_Float16)p;
    p = pv(a0.z, w20.z); lsum += p; P.e[2]  = (_Float16)p;
    p = pv(a0.w, w20.w); lsum += p; P.e[3]  = (_Float16)p;
    p = pv(a1.x, w21.x); lsum += p; P.e[4]  = (_Float16)p;
    p = pv(a1.y, w21.y); lsum += p; P.e[5]  = (_Float16)p;
    p = pv(a1.z, w21.z); lsum += p; P.e[6]  = (_Float16)p;
    p = pv(a1.w, w21.w); lsum += p; P.e[7]  = (_Float16)p;
    p = pv(a2.x, w22.x); lsum += p; P.e[8]  = (_Float16)p;
    p = pv(a2.y, w22.y); lsum += p; P.e[9]  = (_Float16)p;
    p = pv(a2.z, w22.z); lsum += p; P.e[10] = (_Float16)p;
    p = pv(a2.w, w22.w); lsum += p; P.e[11] = (_Float16)p;
    p = pv(a3.x, w23.x); lsum += p; P.e[12] = (_Float16)p;
    p = pv(a3.y, w23.y); lsum += p; P.e[13] = (_Float16)p;
    p = pv(a3.z, w23.z); lsum += p; P.e[14] = (_Float16)p;
    p = pv(a3.w, w23.w); lsum += p; P.e[15] = (_Float16)p;

    #pragma unroll
    for (int nt = 0; nt < 4; ++nt) {
      const _Float16* bp = &ldsB[buf][(nt * 16 + lm) * 40 + half * 8];
      BFrag B;
      B.h[0] = *(const v8h*)bp;
      B.h[1] = *(const v8h*)(bp + 16);
      c[nt].v = __builtin_amdgcn_wmma_f32_16x16x32_f16(
          false, P.v, false, B.v, (short)0, c[nt].v, false, false);
    }
  }

  // finish row sum: lanes l and l^16 hold the two K-halves of the same row
  lsum += __shfl_xor(lsum, 16, 32);

  float* pa = pacc + ((long)blockIdx.y * N_NODES + row0) * OUT_F;
  #pragma unroll
  for (int nt = 0; nt < 4; ++nt)
    #pragma unroll
    for (int r = 0; r < 8; ++r)
      pa[(wave * 16 + r + half * 8) * OUT_F + nt * 16 + lm] = c[nt].e[r];
  if (l < 16)
    pl[(long)blockIdx.y * N_NODES + row0 + wave * 16 + lm] = lsum;
}

// ---------------------------------------------------------------------------
// Kernel 3: reduce j-split partials, normalize (softmax denom), ELU, write out.
// ---------------------------------------------------------------------------
__global__ __launch_bounds__(256)
void gat_reduce(const float* __restrict__ pacc, const float* __restrict__ pl,
                float* __restrict__ out) {
  const int tid = blockIdx.x * 256 + threadIdx.x;   // row*64 + col
  const int row = tid >> 6;
  float acc = 0.0f, ls = 0.0f;
  #pragma unroll
  for (int s = 0; s < NSPLIT; ++s) {
    acc += pacc[(long)s * N_NODES * OUT_F + tid];
    ls  += pl[(long)s * N_NODES + row];
  }
  const float v = acc / ls;
  out[tid] = v > 0.0f ? v : (__expf(v) - 1.0f);     // ELU(alpha=1)
}

// ---------------------------------------------------------------------------
extern "C" void kernel_launch(void* const* d_in, const int* in_sizes, int n_in,
                              void* d_out, int out_size, void* d_ws, size_t ws_size,
                              hipStream_t stream) {
  (void)in_sizes; (void)n_in; (void)out_size; (void)ws_size;
  const float* h   = (const float*)d_in[0];
  const int*   adj = (const int*)d_in[1];
  const float* W   = (const float*)d_in[2];
  const float* a   = (const float*)d_in[3];
  float* out = (float*)d_out;

  char* ws = (char*)d_ws;
  _Float16* WhT = (_Float16*)(ws);                              // 1 MB
  float*    Wh  = (float*)(ws + (1u << 20));                    // 2 MB
  float*    wh1 = (float*)(ws + 3u * (1u << 20));               // 32 KB
  float*    wh2 = (float*)(ws + 3u * (1u << 20) + (32u << 10)); // 32 KB
  float*    pacc= (float*)(ws + 4u * (1u << 20));               // 8 MB
  float*    pl  = (float*)(ws + 12u * (1u << 20));              // 128 KB

  gat_wh_gemm<<<dim3(N_NODES / 64), dim3(128), 0, stream>>>(h, W, Wh, WhT);
  gat_wh12<<<dim3(N_NODES / 256), dim3(256), 0, stream>>>(Wh, a, wh1, wh2);
  gat_attn<<<dim3(N_NODES / 64, NSPLIT), dim3(128), 0, stream>>>(adj, WhT, wh1, wh2, pacc, pl);
  gat_reduce<<<dim3(N_NODES * OUT_F / 256), dim3(256), 0, stream>>>(pacc, pl, out);
}
